// Attention_41678362640976
// MI455X (gfx1250) — matrix operand order repaired and checked by reading, not compiled, not run on hardware
//
#include <hip/hip_runtime.h>
#include <hip/hip_bf16.h>

// ---------- constants ----------
#define BB 8
#define NN 1025
#define DD 768
#define HH 12
#define HD 64
#define MM (BB * NN)          // 8200 rows
#define NVP 1056              // padded key stride for V^T (33*32, 32B-aligned rows)
#define SCALE 0.03608439182435161f   // 768^-0.5 (reference scales by full dim)

typedef unsigned short ushort_t;
typedef __attribute__((ext_vector_type(16))) __bf16 v16bf;
typedef __attribute__((ext_vector_type(8)))  float  v8f;

union BF16x16 { v16bf v; uint4 q[2]; ushort_t u[16]; };

__device__ __forceinline__ ushort_t f2bf(float f) {
    union { float f; unsigned u; } c; c.f = f;
    unsigned u = c.u;
    unsigned r = (u + 0x7FFFu + ((u >> 16) & 1u)) >> 16;
    return (ushort_t)r;
}
__device__ __forceinline__ float bf2f(ushort_t h) {
    union { unsigned u; float f; } c; c.u = ((unsigned)h) << 16;
    return c.f;
}

__device__ __forceinline__ float block_red(float v, float* red, int t, int op_max) {
    red[t] = v; __syncthreads();
    for (int s = 128; s > 0; s >>= 1) {
        if (t < s) red[t] = op_max ? fmaxf(red[t], red[t + s]) : (red[t] + red[t + s]);
        __syncthreads();
    }
    float r = red[0]; __syncthreads();
    return r;
}

// ---------- 0) zero-fill (for V^T padding columns) ----------
__global__ void zero_u128_kernel(uint4* __restrict__ p, long n16) {
    long i = (long)blockIdx.x * blockDim.x + threadIdx.x;
    long stride = (long)gridDim.x * blockDim.x;
    uint4 z; z.x = z.y = z.z = z.w = 0u;
    for (; i < n16; i += stride) p[i] = z;
}

// ---------- 1) LayerNorm + bf16 cast ----------
__global__ void ln_bf16_kernel(const float* __restrict__ x,
                               const float* __restrict__ gw,
                               const float* __restrict__ gb,
                               ushort_t* __restrict__ xn) {
    __shared__ float red[256];
    int row = blockIdx.x;
    int t = threadIdx.x;
    const float* xr = x + (size_t)row * DD;
    float s = 0.f, ss = 0.f;
    for (int i = t; i < DD; i += 256) { float v = xr[i]; s += v; ss += v * v; }
    s  = block_red(s,  red, t, 0);
    ss = block_red(ss, red, t, 0);
    float mu  = s * (1.0f / DD);
    float var = ss * (1.0f / DD) - mu * mu;
    float inv = rsqrtf(var + 1e-5f);
    for (int i = t; i < DD; i += 256) {
        float v = (xr[i] - mu) * inv * gw[i] + gb[i];
        xn[(size_t)row * DD + i] = f2bf(v);
    }
}

// ---------- 2) weights fp32 -> bf16, TRANSPOSED to [N][K] ----------
__global__ void cvt_wT_kernel(const float* __restrict__ wqkv, const float* __restrict__ wout,
                              ushort_t* __restrict__ wqkvT, ushort_t* __restrict__ woutT) {
    const long n1 = (long)(3 * DD) * DD;
    const long n2 = (long)DD * DD;
    long id = (long)blockIdx.x * blockDim.x + threadIdx.x;
    long stride = (long)gridDim.x * blockDim.x;
    for (long j = id; j < n1; j += stride) {
        long n = j / DD, k = j % DD;                  // output [n][k]
        wqkvT[j] = f2bf(wqkv[k * (long)(3 * DD) + n]);
    }
    for (long j = id; j < n2; j += stride) {
        long n = j / DD, k = j % DD;
        woutT[j] = f2bf(wout[k * (long)DD + n]);
    }
}

// ---------- 3/6) LDS-free, software-pipelined WMMA GEMM ----------
// C[M,Nc] = A[M,K] * B[K,Nc], B given K-major ([Ntot][Ktot]).
// mode 0: scatter Q,K -> [B,H,N,HD] bf16, V -> [B,H,HD,NVP] bf16 (transposed).
// mode 1: C = acc + bias (fp32 contiguous).
__global__ void gemm_wmma_kernel(const ushort_t* __restrict__ A,
                                 const ushort_t* __restrict__ BT,
                                 int Mtot, int Ntot, int Ktot,
                                 ushort_t* __restrict__ Qo, ushort_t* __restrict__ Ko,
                                 ushort_t* __restrict__ Vo,
                                 float* __restrict__ Co, const float* __restrict__ bias,
                                 int mode) {
    const int n0 = blockIdx.x * 64;
    const int m0 = blockIdx.y * 64;
    const int tid = threadIdx.x;           // 128 threads = 4 waves
    const int w = tid >> 5, L = tid & 31;
    const int L16 = L & 15, Lh = L >> 4;

    int gm = m0 + w * 16 + L16;
    if (gm >= Mtot) gm = Mtot - 1;         // clamped rows masked at store
    const ushort_t* arow = A + (size_t)gm * Ktot;
    const ushort_t* brow[4];
#pragma unroll
    for (int nb = 0; nb < 4; ++nb) {
        int gn = n0 + nb * 16 + L16;
        if (gn >= Ntot) gn = Ntot - 1;
        brow[nb] = BT + (size_t)gn * Ktot;
    }

    v8f acc[4];
#pragma unroll
    for (int i = 0; i < 4; ++i) acc[i] = (v8f){0,0,0,0,0,0,0,0};

    // prologue: fragments for k0 = 0
    BF16x16 af, bf[4];
    af.q[0] = *(const uint4*)(arow + Lh * 16);
    af.q[1] = *(const uint4*)(arow + Lh * 16 + 8);
#pragma unroll
    for (int nb = 0; nb < 4; ++nb) {
        bf[nb].q[0] = *(const uint4*)(brow[nb] + Lh * 16);
        bf[nb].q[1] = *(const uint4*)(brow[nb] + Lh * 16 + 8);
    }

#pragma unroll 2
    for (int k0 = 0; k0 < Ktot; k0 += 32) {
        const int kn = (k0 + 32 < Ktot) ? k0 + 32 : k0;   // last iter reloads (discarded)
        BF16x16 afn, bfn[4];
        afn.q[0] = *(const uint4*)(arow + kn + Lh * 16);
        afn.q[1] = *(const uint4*)(arow + kn + Lh * 16 + 8);
#pragma unroll
        for (int nb = 0; nb < 4; ++nb) {
            bfn[nb].q[0] = *(const uint4*)(brow[nb] + kn + Lh * 16);
            bfn[nb].q[1] = *(const uint4*)(brow[nb] + kn + Lh * 16 + 8);
        }
        __builtin_prefetch(arow + k0 + 64, 0, 1);
#pragma unroll
        for (int nb = 0; nb < 4; ++nb)
            acc[nb] = __builtin_amdgcn_wmma_f32_16x16x32_bf16(
                false, af.v, false, bf[nb].v, (short)0, acc[nb], false, false);
        af = afn;
#pragma unroll
        for (int nb = 0; nb < 4; ++nb) bf[nb] = bfn[nb];
    }

#pragma unroll
    for (int nb = 0; nb < 4; ++nb) {
#pragma unroll
        for (int r = 0; r < 8; ++r) {
            int row = m0 + w * 16 + r + 8 * Lh;     // C layout: M = vgpr + 8*(lane/16)
            int col = n0 + nb * 16 + L16;
            if (row < Mtot && col < Ntot) {
                float v = acc[nb][r];
                if (mode == 0) {
                    int which = col / DD, rem = col % DD;
                    int h = rem >> 6, dd = rem & 63;
                    int b = row / NN, n = row % NN;
                    ushort_t bv = f2bf(v);
                    if (which == 0)
                        Qo[((size_t)(b * HH + h) * NN + n) * HD + dd] = bv;
                    else if (which == 1)
                        Ko[((size_t)(b * HH + h) * NN + n) * HD + dd] = bv;
                    else
                        Vo[((size_t)(b * HH + h) * HD + dd) * NVP + n] = bv;  // transposed
                } else {
                    Co[(size_t)row * Ntot + col] = v + bias[col];
                }
            }
        }
    }
}

// ---------- 4) flash attention, barrier-free; operand frags straight from global ----------
__global__ void flash_attn_kernel(const ushort_t* __restrict__ Q,
                                  const ushort_t* __restrict__ K,
                                  const ushort_t* __restrict__ Vt,
                                  ushort_t* __restrict__ AO) {
    __shared__ ushort_t sP[4][16][40];      // per-wave P bounce (C-layout -> A-layout)

    const int bh = blockIdx.y;
    const int b = bh / HH, h = bh % HH;
    const int tid = threadIdx.x;            // 128 threads = 4 independent waves
    const int w = tid >> 5, L = tid & 31;
    const int L16 = L & 15, Lh = L >> 4;
    const int mtile = blockIdx.x * 4 + w;   // 65 valid tiles of 16 rows
    const bool active = (mtile < 65);
    const int mbase = mtile * 16;

    const ushort_t* Qb  = Q  + (size_t)bh * NN * HD;
    const ushort_t* Kb  = K  + (size_t)bh * NN * HD;
    const ushort_t* Vtb = Vt + (size_t)bh * HD * NVP;
    const ushort_t* vrow[4];
#pragma unroll
    for (int nb = 0; nb < 4; ++nb)
        vrow[nb] = Vtb + (size_t)(nb * 16 + L16) * NVP;

    int qrow = mbase + L16;
    if (!active || qrow >= NN) qrow = 0;
    BF16x16 qf[2];
#pragma unroll
    for (int kc = 0; kc < 2; ++kc) {
        qf[kc].q[0] = *(const uint4*)(Qb + (size_t)qrow * HD + kc * 32 + Lh * 16);
        qf[kc].q[1] = *(const uint4*)(Qb + (size_t)qrow * HD + kc * 32 + Lh * 16 + 8);
    }

    v8f acc[4];
#pragma unroll
    for (int i = 0; i < 4; ++i) acc[i] = (v8f){0,0,0,0,0,0,0,0};
    float mrun[8], lrun[8];
#pragma unroll
    for (int r = 0; r < 8; ++r) { mrun[r] = -__builtin_inff(); lrun[r] = 0.f; }

    for (int jt = 0; jt < 33; ++jt) {       // 33*32 = 1056 >= 1025 keys
        const int kbase = jt * 32;
        if (!active) continue;

        // ---- 1) issue ALL K-fragment loads (one clause), then S WMMAs ----
        BF16x16 kf[2][2];
#pragma unroll
        for (int blk = 0; blk < 2; ++blk) {
            int key = kbase + blk * 16 + L16;
            int keyc = (key < NN) ? key : (NN - 1);
            const ushort_t* krow = Kb + (size_t)keyc * HD;
#pragma unroll
            for (int kc = 0; kc < 2; ++kc) {
                kf[blk][kc].q[0] = *(const uint4*)(krow + kc * 32 + Lh * 16);
                kf[blk][kc].q[1] = *(const uint4*)(krow + kc * 32 + Lh * 16 + 8);
            }
        }
        v8f s[2];
        s[0] = (v8f){0,0,0,0,0,0,0,0};
        s[1] = (v8f){0,0,0,0,0,0,0,0};
#pragma unroll
        for (int blk = 0; blk < 2; ++blk)
#pragma unroll
            for (int kc = 0; kc < 2; ++kc)
                s[blk] = __builtin_amdgcn_wmma_f32_16x16x32_bf16(
                    false, qf[kc].v, false, kf[blk][kc].v, (short)0, s[blk], false, false);

        // ---- 2) issue V-fragment loads NOW; they fly during the softmax VALU phase ----
        BF16x16 vf[4];
#pragma unroll
        for (int nb = 0; nb < 4; ++nb) {
            vf[nb].q[0] = *(const uint4*)(vrow[nb] + kbase + Lh * 16);
            vf[nb].q[1] = *(const uint4*)(vrow[nb] + kbase + Lh * 16 + 8);
        }

        // ---- 3) scale + invalid-key mask ----
#pragma unroll
        for (int blk = 0; blk < 2; ++blk) {
            int key = kbase + blk * 16 + L16;
            bool bad = (key >= NN);
#pragma unroll
            for (int r = 0; r < 8; ++r)
                s[blk][r] = bad ? -1e30f : s[blk][r] * SCALE;
        }
        // ---- 4) online softmax (row stats across the 16-lane group) ----
        float pmax[8];
#pragma unroll
        for (int r = 0; r < 8; ++r) pmax[r] = fmaxf(s[0][r], s[1][r]);
#pragma unroll
        for (int m = 1; m <= 8; m <<= 1)
#pragma unroll
            for (int r = 0; r < 8; ++r)
                pmax[r] = fmaxf(pmax[r], __shfl_xor(pmax[r], m, 32));
        float alpha[8];
#pragma unroll
        for (int r = 0; r < 8; ++r) {
            float mn = fmaxf(mrun[r], pmax[r]);
            alpha[r] = __expf(mrun[r] - mn);
            mrun[r] = mn;
        }
        float rsum[8];
#pragma unroll
        for (int blk = 0; blk < 2; ++blk)
#pragma unroll
            for (int r = 0; r < 8; ++r)
                s[blk][r] = __expf(s[blk][r] - mrun[r]);
#pragma unroll
        for (int r = 0; r < 8; ++r) rsum[r] = s[0][r] + s[1][r];
#pragma unroll
        for (int m = 1; m <= 8; m <<= 1)
#pragma unroll
            for (int r = 0; r < 8; ++r)
                rsum[r] += __shfl_xor(rsum[r], m, 32);
#pragma unroll
        for (int r = 0; r < 8; ++r) lrun[r] = lrun[r] * alpha[r] + rsum[r];
#pragma unroll
        for (int nb = 0; nb < 4; ++nb)
#pragma unroll
            for (int r = 0; r < 8; ++r) acc[nb][r] *= alpha[r];

        // ---- 5) P: C-layout -> A-layout via per-wave LDS slice ----
#pragma unroll
        for (int blk = 0; blk < 2; ++blk)
#pragma unroll
            for (int r = 0; r < 8; ++r)
                sP[w][r + 8 * Lh][blk * 16 + L16] = f2bf(s[blk][r]);

        BF16x16 pf;
        pf.q[0] = *(const uint4*)&sP[w][L16][Lh * 16];
        pf.q[1] = *(const uint4*)&sP[w][L16][Lh * 16 + 8];

        // ---- 6) O += P @ V (V-frags already in flight/registers) ----
#pragma unroll
        for (int nb = 0; nb < 4; ++nb)
            acc[nb] = __builtin_amdgcn_wmma_f32_16x16x32_bf16(
                false, pf.v, false, vf[nb].v, (short)0, acc[nb], false, false);
    }

    if (active) {
#pragma unroll
        for (int nb = 0; nb < 4; ++nb)
#pragma unroll
            for (int r = 0; r < 8; ++r) {
                int row = mbase + r + 8 * Lh;
                if (row < NN) {
                    float o = acc[nb][r] / lrun[r];
                    int col = h * HD + nb * 16 + L16;
                    AO[(size_t)(b * NN + row) * DD + col] = f2bf(o);
                }
            }
    }
}

// ---------- 5) exact CLS-row (row 0) with prior injection ----------
__global__ void cls_row_kernel(const ushort_t* __restrict__ Q,
                               const ushort_t* __restrict__ K,
                               const ushort_t* __restrict__ Vt,
                               const float* __restrict__ canny,
                               const float* __restrict__ noise,
                               ushort_t* __restrict__ AO) {
    __shared__ float dots[1056];
    __shared__ float q0s[HD];
    __shared__ float red[256];
    const int bh = blockIdx.x;
    const int b = bh / HH, h = bh % HH;
    const int t = threadIdx.x;
    const size_t kbase = (size_t)bh * NN * HD;

    if (t < HD) q0s[t] = bf2f(Q[kbase + t]);
    __syncthreads();

    for (int j = t; j < NN; j += 256) {
        float d = 0.f;
#pragma unroll 8
        for (int k = 0; k < HD; ++k) d += bf2f(K[kbase + (size_t)j * HD + k]) * q0s[k];
        dots[j] = d * SCALE;
    }
    __syncthreads();

    // softmax over patch keys (j = 1..N-1)
    float mx = -__builtin_inff();
    for (int j = t; j < NN; j += 256) if (j >= 1) mx = fmaxf(mx, dots[j]);
    mx = block_red(mx, red, t, 1);
    float es = 0.f;
    for (int j = t; j < NN; j += 256) if (j >= 1) es += __expf(dots[j] - mx);
    es = block_red(es, red, t, 0);

    float cs = 0.f, ns = 0.f;
    for (int j = t; j < 1024; j += 256) {
        cs += canny[b * 1024 + j] + 1.0f;
        ns += noise[b * 1024 + j];
    }
    cs = block_red(cs, red, t, 0);
    ns = block_red(ns, red, t, 0);

    for (int j = t; j < NN; j += 256) if (j >= 1) {
        float p = __expf(dots[j] - mx) / es;
        dots[j] = p + (canny[b * 1024 + j - 1] + 1.0f) / cs + noise[b * 1024 + j - 1] / ns;
    }
    __syncthreads();

    // second (full) softmax over all N entries (entry 0 keeps raw scaled dot)
    float mx2 = -__builtin_inff();
    for (int j = t; j < NN; j += 256) mx2 = fmaxf(mx2, dots[j]);
    mx2 = block_red(mx2, red, t, 1);
    float es2 = 0.f;
    for (int j = t; j < NN; j += 256) es2 += __expf(dots[j] - mx2);
    es2 = block_red(es2, red, t, 0);
    for (int j = t; j < NN; j += 256) dots[j] = __expf(dots[j] - mx2) / es2;
    __syncthreads();

    if (t < HD) {
        const ushort_t* vrow = Vt + ((size_t)bh * HD + t) * NVP;
        float o = 0.f;
        for (int j = 0; j < NN; ++j) o += dots[j] * bf2f(vrow[j]);
        AO[(size_t)(b * NN) * DD + h * HD + t] = f2bf(o);
    }
}

// ---------- launch ----------
extern "C" void kernel_launch(void* const* d_in, const int* in_sizes, int n_in,
                              void* d_out, int out_size, void* d_ws, size_t ws_size,
                              hipStream_t stream) {
    const float* x      = (const float*)d_in[0];
    const float* canny  = (const float*)d_in[1];
    const float* noise  = (const float*)d_in[2];
    const float* ln_w   = (const float*)d_in[3];
    const float* ln_b   = (const float*)d_in[4];
    const float* w_qkv  = (const float*)d_in[5];
    const float* w_out  = (const float*)d_in[6];
    const float* b_out  = (const float*)d_in[7];
    float* out = (float*)d_out;

    char* ws = (char*)d_ws;
    size_t off = 0;
    auto take = [&](size_t bytes) -> char* {
        char* p = ws + off;
        off = (off + bytes + 255) & ~(size_t)255;
        return p;
    };
    const size_t qk_elems = (size_t)BB * HH * NN * HD;    // 6,297,600
    const size_t vt_elems = (size_t)BB * HH * HD * NVP;   // 6,488,064
    ushort_t* xn      = (ushort_t*)take((size_t)MM * DD * 2);
    ushort_t* wqkvT   = (ushort_t*)take((size_t)DD * 3 * DD * 2);
    ushort_t* woutT   = (ushort_t*)take((size_t)DD * DD * 2);
    ushort_t* Qb      = (ushort_t*)take(qk_elems * 2);
    ushort_t* Kb      = (ushort_t*)take(qk_elems * 2);
    ushort_t* Vtb     = (ushort_t*)take(vt_elems * 2);
    ushort_t* ao      = (ushort_t*)take((size_t)MM * DD * 2);
    (void)ws_size; (void)in_sizes; (void)n_in; (void)out_size;

    ln_bf16_kernel<<<MM, 256, 0, stream>>>(x, ln_w, ln_b, xn);
    cvt_wT_kernel<<<2048, 256, 0, stream>>>(w_qkv, w_out, wqkvT, woutT);
    zero_u128_kernel<<<1024, 256, 0, stream>>>((uint4*)Vtb, (long)(vt_elems / 8));

    // QKV projection: [8200,768] x [768,2304] (B K-major)
    gemm_wmma_kernel<<<dim3((3 * DD) / 64, (MM + 63) / 64), 128, 0, stream>>>(
        xn, wqkvT, MM, 3 * DD, DD, Qb, Kb, Vtb, nullptr, nullptr, 0);

    // flash attention, 65 row-tiles of 16, 4 waves per block, per (b,h)
    flash_attn_kernel<<<dim3(17, BB * HH), 128, 0, stream>>>(Qb, Kb, Vtb, ao);

    // exact CLS row with prior injection (overwrites row 0 of ao)
    cls_row_kernel<<<BB * HH, 256, 0, stream>>>(Qb, Kb, Vtb, canny, noise, ao);

    // output projection + bias: [8200,768] x [768,768] -> fp32 out
    gemm_wmma_kernel<<<dim3(DD / 64, (MM + 63) / 64), 128, 0, stream>>>(
        ao, woutT, MM, DD, DD, nullptr, nullptr, nullptr, out, b_out, 1);
}
